// TransformerBlock_54674933678394
// MI455X (gfx1250) — matrix operand order repaired and checked by reading, not compiled, not run on hardware
//
#include <hip/hip_runtime.h>

// ---------------------------------------------------------------------------
// MI455X (gfx1250) CDNA5 implementation: bf16 WMMA (16x16x32, f32 accum)
// for all GEMMs and attention; wave32 throughout. Attention Q/K staging uses
// GLOBAL_LOAD_ASYNC_TO_LDS_B128 (ASYNCcnt) per CDNA5 ISA §15.18.3.
// ---------------------------------------------------------------------------

#define E_DIM 512
#define SEQ   80
#define NBAT  256
#define NH    8
#define HD    64

typedef __attribute__((ext_vector_type(16))) __bf16 v16bf;
typedef __attribute__((ext_vector_type(8)))  __bf16 v8bf;
typedef __attribute__((ext_vector_type(8)))  float  v8f;

static __device__ inline v16bf frag_cat(v8bf lo, v8bf hi) {
  union { v8bf h[2]; v16bf f; } u;
  u.h[0] = lo; u.h[1] = hi;
  return u.f;
}

static __device__ inline v8f wmma_bf16(v16bf a, v16bf b, v8f c) {
  // 8 args: (neg_a, A, neg_b, B, c_mod, C, reuse_a, reuse_b)
  return __builtin_amdgcn_wmma_f32_16x16x32_bf16(false, a, false, b, (short)0, c,
                                                 false, false);
}

// ---------------------------------------------------------------------------
// Generic GEMM: Out[m,n] = act( sum_k A[m,k] * W[n,k] + bias[n] )
//   A: (M,K) f32 or bf16 row-major; W: (N,K) bf16 row-major (pre-converted)
//   block = 256 threads (8 waves); wave computes 32(M) x 64(N)
//   grid = (M/256, N/64); M % 256 == 0, N % 64 == 0, K % 32 == 0
// ---------------------------------------------------------------------------
template <bool ABF16, bool RELU, bool OUTBF16>
__global__ __launch_bounds__(256) void gemm_wmma(
    const void* __restrict__ Ap, const __bf16* __restrict__ W,
    const float* __restrict__ bias, void* __restrict__ Outp,
    int M, int N, int K)
{
  const int lane = threadIdx.x & 31;
  const int wave = threadIdx.x >> 5;
  const int half = lane >> 4;        // 0: lanes 0-15, 1: lanes 16-31
  const int l15  = lane & 15;
  const int mbase = blockIdx.x * 256 + wave * 32;
  const int col0  = blockIdx.y * 64;

  v8f acc[2][4] = {};
  const int aoff = half * 16;        // A: lanes>=16 hold K = 16..31

  for (int k0 = 0; k0 < K; k0 += 32) {
    v16bf afrag[2];
#pragma unroll
    for (int t = 0; t < 2; ++t) {
      const int row = mbase + t * 16 + l15;   // A-fragment row (M = lane&15)
      if constexpr (ABF16) {
        const __bf16* Arow = (const __bf16*)Ap + (size_t)row * K + k0 + aoff;
        afrag[t] = frag_cat(*(const v8bf*)(Arow), *(const v8bf*)(Arow + 8));
        if (k0 + 32 < K) __builtin_prefetch(Arow + 32, 0, 0);
      } else {
        const float* Arow = (const float*)Ap + (size_t)row * K + k0 + aoff;
        const float4 a0 = *(const float4*)(Arow + 0);
        const float4 a1 = *(const float4*)(Arow + 4);
        const float4 a2 = *(const float4*)(Arow + 8);
        const float4 a3 = *(const float4*)(Arow + 12);
        v16bf t16;
        t16[0]=(__bf16)a0.x; t16[1]=(__bf16)a0.y; t16[2]=(__bf16)a0.z; t16[3]=(__bf16)a0.w;
        t16[4]=(__bf16)a1.x; t16[5]=(__bf16)a1.y; t16[6]=(__bf16)a1.z; t16[7]=(__bf16)a1.w;
        t16[8]=(__bf16)a2.x; t16[9]=(__bf16)a2.y; t16[10]=(__bf16)a2.z; t16[11]=(__bf16)a2.w;
        t16[12]=(__bf16)a3.x; t16[13]=(__bf16)a3.y; t16[14]=(__bf16)a3.z; t16[15]=(__bf16)a3.w;
        afrag[t] = t16;
        if (k0 + 32 < K) __builtin_prefetch(Arow + 32, 0, 0);
      }
    }
#pragma unroll
    for (int j = 0; j < 4; ++j) {
      // B element e <-> K = k0 + 16*half + e ; N = col0 + j*16 + (lane&15)
      const __bf16* Wrow = W + (size_t)(col0 + j * 16 + l15) * K + k0 + half * 16;
      const v16bf bfrag = *(const v16bf*)Wrow;
      acc[0][j] = wmma_bf16(afrag[0], bfrag, acc[0][j]);
      acc[1][j] = wmma_bf16(afrag[1], bfrag, acc[1][j]);
    }
  }

#pragma unroll
  for (int t = 0; t < 2; ++t) {
#pragma unroll
    for (int j = 0; j < 4; ++j) {
      const int col = col0 + j * 16 + l15;
      const float bv = bias ? bias[col] : 0.0f;
#pragma unroll
      for (int r = 0; r < 8; ++r) {
        const int m = mbase + t * 16 + r + half * 8;  // C: M = r + 8*(lane>=16)
        float v = acc[t][j][r] + bv;
        if (RELU) v = v > 0.0f ? v : 0.0f;
        if constexpr (OUTBF16)
          ((__bf16*)Outp)[(size_t)m * N + col] = (__bf16)v;
        else
          ((float*)Outp)[(size_t)m * N + col] = v;
      }
    }
  }
}

// ---------------------------------------------------------------------------
// Attention: one workgroup (5 waves / 160 threads) per (n, h).
//   Q,K,V: bf16 in (n,s,h,d) layout; O: f32 same layout.
//   Q/K tiles staged to LDS via GLOBAL_LOAD_ASYNC_TO_LDS_B128 (ASYNCcnt),
//   V transposed into LDS on the VALU path.
//   S = scale * Q K^T (5x5 WMMA tiles, Kdim=64); softmax in registers;
//   O = P V (K padded 80->96 with zeros).
// ---------------------------------------------------------------------------
__global__ __launch_bounds__(160) void attn_wmma(
    const __bf16* __restrict__ Q, const __bf16* __restrict__ Kg,
    const __bf16* __restrict__ V, float* __restrict__ O, float scale)
{
  __shared__ __align__(32) __bf16 sQ[SEQ * HD];     // 10240 B
  __shared__ __align__(32) __bf16 sK[SEQ * HD];     // 10240 B
  __shared__ __align__(32) __bf16 sVt[HD * 96];     // 12288 B (transposed, padded)
  __shared__ __align__(32) __bf16 sP[SEQ * 96];     // 15360 B (padded)

  const int nb = blockIdx.x;         // n*8 + h
  const int n = nb >> 3, h = nb & 7;
  const size_t base = ((size_t)n * (SEQ * NH) + h) * HD;  // elem offset; +s*512+d
  const int tid = threadIdx.x;

  // ---- async staging of Q and K: 640 x 16B chunks each, 4 full passes ----
  {
    const unsigned ldsQ = (unsigned)(unsigned long long)(void*)sQ;
    const unsigned ldsK = (unsigned)(unsigned long long)(void*)sK;
    const unsigned long long gQ = (unsigned long long)Q;
    const unsigned long long gK = (unsigned long long)Kg;
#pragma unroll
    for (int p = 0; p < 4; ++p) {
      const int c = p * 160 + tid;                 // chunk of 8 bf16
      const int s = c >> 3, cd = (c & 7) * 8;
      const unsigned goff = (unsigned)((base + (size_t)s * (NH * HD) + cd) * 2);
      const unsigned lq = ldsQ + (unsigned)c * 16u;
      const unsigned lk = ldsK + (unsigned)c * 16u;
      asm volatile("global_load_async_to_lds_b128 %0, %1, %2"
                   :: "v"(lq), "v"(goff), "s"(gQ) : "memory");
      asm volatile("global_load_async_to_lds_b128 %0, %1, %2"
                   :: "v"(lk), "v"(goff), "s"(gK) : "memory");
    }
  }
  // ---- V transposed into LDS (+ zero pad rows 80..95) ----
  for (int i = tid; i < SEQ * HD; i += 160) {
    const int s = i >> 6, d = i & 63;
    sVt[d * 96 + s] = V[base + (size_t)s * (NH * HD) + d];
  }
  for (int i = tid; i < HD * 16; i += 160) {
    const int d = i >> 4, s = SEQ + (i & 15);
    sVt[d * 96 + s] = (__bf16)0.0f;
  }
  asm volatile("s_wait_asynccnt 0x0" ::: "memory");
  __syncthreads();

  const int lane = tid & 31, wave = tid >> 5;
  const int half = lane >> 4, l15 = lane & 15;
  const int m0 = wave * 16;          // this wave's query-row block

  // ---- S = Q K^T ----
  v8f sacc[5] = {};
  for (int k0 = 0; k0 < HD; k0 += 32) {
    const __bf16* qr = &sQ[(m0 + l15) * HD + k0 + half * 16];
    const v16bf af = frag_cat(*(const v8bf*)qr, *(const v8bf*)(qr + 8));
#pragma unroll
    for (int j = 0; j < 5; ++j) {
      const __bf16* kr = &sK[(j * 16 + l15) * HD + k0 + half * 16];
      sacc[j] = wmma_bf16(af, *(const v16bf*)kr, sacc[j]);
    }
  }

  // ---- softmax over each row (row data lives in one 16-lane half) ----
#pragma unroll
  for (int r = 0; r < 8; ++r) {
    float x[5];
    float mx = -1e30f;
#pragma unroll
    for (int j = 0; j < 5; ++j) { x[j] = sacc[j][r] * scale; mx = fmaxf(mx, x[j]); }
    for (int off = 1; off < 16; off <<= 1) mx = fmaxf(mx, __shfl_xor(mx, off, 32));
    float sum = 0.0f;
#pragma unroll
    for (int j = 0; j < 5; ++j) { x[j] = __expf(x[j] - mx); sum += x[j]; }
    for (int off = 1; off < 16; off <<= 1) sum += __shfl_xor(sum, off, 32);
    const float inv = 1.0f / sum;
    const int mrow = m0 + r + half * 8;
#pragma unroll
    for (int j = 0; j < 5; ++j) sP[mrow * 96 + j * 16 + l15] = (__bf16)(x[j] * inv);
    sP[mrow * 96 + 80 + l15] = (__bf16)0.0f;   // zero-pad cols 80..95
  }
  __syncthreads();

  // ---- O = P V  (K = 96) ----
  v8f oacc[4] = {};
  for (int k0 = 0; k0 < 96; k0 += 32) {
    const __bf16* pr = &sP[(m0 + l15) * 96 + k0 + half * 16];
    const v16bf af = frag_cat(*(const v8bf*)pr, *(const v8bf*)(pr + 8));
#pragma unroll
    for (int j = 0; j < 4; ++j) {
      const __bf16* vr = &sVt[(j * 16 + l15) * 96 + k0 + half * 16];
      oacc[j] = wmma_bf16(af, *(const v16bf*)vr, oacc[j]);
    }
  }
#pragma unroll
  for (int j = 0; j < 4; ++j) {
#pragma unroll
    for (int r = 0; r < 8; ++r) {
      const int s = m0 + r + half * 8;
      const int d = j * 16 + l15;
      O[base + (size_t)s * (NH * HD) + d] = oacc[j][r];
    }
  }
}

// ---------------------------------------------------------------------------
// Fused residual-add + LayerNorm: Y = LN(X + R) * g + b ; one wave per row.
// ---------------------------------------------------------------------------
__global__ __launch_bounds__(256) void add_ln(
    const float* __restrict__ X, const float* __restrict__ R,
    const float* __restrict__ g, const float* __restrict__ b,
    float* __restrict__ Y)
{
  const int row  = blockIdx.x * 8 + (threadIdx.x >> 5);
  const int lane = threadIdx.x & 31;
  const size_t off = (size_t)row * E_DIM + lane * 16;
  float v[16];
  float s = 0.0f, ss = 0.0f;
#pragma unroll
  for (int q = 0; q < 4; ++q) {
    const float4 x = ((const float4*)(X + off))[q];
    const float4 r = ((const float4*)(R + off))[q];
    v[q*4+0] = x.x + r.x; v[q*4+1] = x.y + r.y;
    v[q*4+2] = x.z + r.z; v[q*4+3] = x.w + r.w;
  }
#pragma unroll
  for (int c = 0; c < 16; ++c) { s += v[c]; ss += v[c] * v[c]; }
  for (int o = 1; o < 32; o <<= 1) { s += __shfl_xor(s, o, 32); ss += __shfl_xor(ss, o, 32); }
  const float mean = s * (1.0f / E_DIM);
  const float var  = ss * (1.0f / E_DIM) - mean * mean;
  const float rstd = rsqrtf(var + 1e-5f);
#pragma unroll
  for (int q = 0; q < 4; ++q) {
    float4 o;
    const int c0 = lane * 16 + q * 4;
    o.x = (v[q*4+0] - mean) * rstd * g[c0+0] + b[c0+0];
    o.y = (v[q*4+1] - mean) * rstd * g[c0+1] + b[c0+1];
    o.z = (v[q*4+2] - mean) * rstd * g[c0+2] + b[c0+2];
    o.w = (v[q*4+3] - mean) * rstd * g[c0+3] + b[c0+3];
    ((float4*)(Y + off))[q] = o;
  }
}

// ---------------------------------------------------------------------------
// Small utility kernels
// ---------------------------------------------------------------------------
__global__ void cvt_f32_bf16(const float* __restrict__ s, __bf16* __restrict__ d, int n) {
  const int i = blockIdx.x * 256 + threadIdx.x;
  if (i < n) d[i] = (__bf16)s[i];
}

__global__ void set_scalar(float* p) {
  if (blockIdx.x == 0 && threadIdx.x == 0) *p = 0.0f;
}

// dst (256,80,512) = concat([a(256,5,512), q(256,75,512)], axis=1)
__global__ void build_full(float* __restrict__ dst, const float* __restrict__ a,
                           const float* __restrict__ q) {
  const int i = blockIdx.x * 256 + threadIdx.x;
  if (i >= NBAT * SEQ * E_DIM) return;
  const int c = i & 511;
  const int ns = i >> 9;
  const int n = ns / SEQ, s2 = ns % SEQ;
  dst[i] = (s2 < 5) ? a[((size_t)n * 5 + s2) * E_DIM + c]
                    : q[((size_t)n * 75 + (s2 - 5)) * E_DIM + c];
}

// Xcat (1280,1024) = concat([word, image], axis=-1)
__global__ void build_xcat(float* __restrict__ dst, const float* __restrict__ w,
                           const float* __restrict__ im) {
  const int i = blockIdx.x * 256 + threadIdx.x;
  if (i >= 1280 * 1024) return;
  const int row = i >> 10, c = i & 1023;
  dst[i] = (c < 512) ? w[(size_t)row * 512 + c] : im[(size_t)row * 512 + (c - 512)];
}

// pro[n,j,:] = Z[n,j,:] (j<5) ; test[n,0,:] = Z[n,79,:]
__global__ void slice_out(const float* __restrict__ Z, float* __restrict__ pro,
                          float* __restrict__ test) {
  const int i = blockIdx.x * 256 + threadIdx.x;
  if (i >= NBAT * 6 * E_DIM) return;
  const int c = i & 511;
  const int rj = i >> 9;
  const int n = rj / 6, j = rj % 6;
  if (j < 5) pro[((size_t)n * 5 + j) * E_DIM + c] = Z[((size_t)n * SEQ + j) * E_DIM + c];
  else       test[(size_t)n * E_DIM + c]          = Z[((size_t)n * SEQ + 79) * E_DIM + c];
}

// ---------------------------------------------------------------------------
// Host orchestration
// ---------------------------------------------------------------------------
extern "C" void kernel_launch(void* const* d_in, const int* in_sizes, int n_in,
                              void* d_out, int out_size, void* d_ws, size_t ws_size,
                              hipStream_t stream) {
  (void)in_sizes; (void)n_in; (void)out_size; (void)ws_size;
  const float* word  = (const float*)d_in[0];
  const float* image = (const float*)d_in[1];
  const float* query = (const float*)d_in[2];
  // d_in[3] = s_word (unused by reference), d_in[4] = knovel (== 5)
  const float *Wq[3], *Wk[3], *Wv[3], *W1[3], *b1[3], *W2[3], *b2[3];
  const float *ga[3], *ba[3], *gn[3], *bn[3];
  int wi = 5;
  for (int i = 0; i < 3; ++i) {
    Wq[i] = (const float*)d_in[wi + 0];  Wk[i] = (const float*)d_in[wi + 1];
    Wv[i] = (const float*)d_in[wi + 2];
    W1[i] = (const float*)d_in[wi + 3];  b1[i] = (const float*)d_in[wi + 4];
    W2[i] = (const float*)d_in[wi + 5];  b2[i] = (const float*)d_in[wi + 6];
    ga[i] = (const float*)d_in[wi + 7];  ba[i] = (const float*)d_in[wi + 8];
    gn[i] = (const float*)d_in[wi + 9];  bn[i] = (const float*)d_in[wi + 10];
    wi += 11;
  }
  const float* Wf1 = (const float*)d_in[38];
  const float* bf1 = (const float*)d_in[39];
  const float* Wf2 = (const float*)d_in[40];
  const float* bf2 = (const float*)d_in[41];

  char* pw = (char*)d_ws;
  auto carve = [&](size_t bytes) -> void* {
    void* r = (void*)pw;
    pw += (bytes + 255) & ~(size_t)255;
    return r;
  };
  const size_t MR = (size_t)NBAT * SEQ;  // 20480 rows
  float*  Xf    = (float*) carve(MR * E_DIM * 4);
  float*  Obuf  = (float*) carve(MR * E_DIM * 4);
  float*  Ybuf  = (float*) carve(MR * E_DIM * 4);
  float*  H2    = (float*) carve(MR * E_DIM * 4);
  __bf16* Qb    = (__bf16*)carve(MR * E_DIM * 2);
  __bf16* Kb    = (__bf16*)carve(MR * E_DIM * 2);
  __bf16* Vb    = (__bf16*)carve(MR * E_DIM * 2);
  __bf16* H1    = (__bf16*)carve(MR * E_DIM * 2);
  float*  fused = (float*) carve(1280ull * 512 * 4);
  float*  Xcat  = (float*) carve(1280ull * 1024 * 4);
  __bf16* Hf    = (__bf16*)carve(1280ull * 512 * 2);
  __bf16 *Wqb[3], *Wkb[3], *Wvb[3], *W1b[3], *W2b[3];
  for (int i = 0; i < 3; ++i) {
    Wqb[i] = (__bf16*)carve(64 * 64 * 2);
    Wkb[i] = (__bf16*)carve(64 * 64 * 2);
    Wvb[i] = (__bf16*)carve(64 * 64 * 2);
    W1b[i] = (__bf16*)carve(512 * 512 * 2);
    W2b[i] = (__bf16*)carve(512 * 512 * 2);
  }
  __bf16* Wf1b = (__bf16*)carve(512ull * 1024 * 2);
  __bf16* Wf2b = (__bf16*)carve(512ull * 512 * 2);

  auto cvt = [&](const float* s, __bf16* d, int n) {
    cvt_f32_bf16<<<(n + 255) / 256, 256, 0, stream>>>(s, d, n);
  };
  for (int i = 0; i < 3; ++i) {
    cvt(Wq[i], Wqb[i], 4096); cvt(Wk[i], Wkb[i], 4096); cvt(Wv[i], Wvb[i], 4096);
    cvt(W1[i], W1b[i], 262144); cvt(W2[i], W2b[i], 262144);
  }
  cvt(Wf1, Wf1b, 524288);
  cvt(Wf2, Wf2b, 262144);

  float* out = (float*)d_out;
  set_scalar<<<1, 1, 0, stream>>>(out + 2359296);  // loss = 0

  // Fusion MLP: fused = relu(cat(word,image) @ Wf1^T + bf1) @ Wf2^T + bf2
  build_xcat<<<(1280 * 1024 + 255) / 256, 256, 0, stream>>>(Xcat, word, image);
  gemm_wmma<false, true,  true ><<<dim3(1280 / 256, 512 / 64), 256, 0, stream>>>(
      Xcat, Wf1b, bf1, Hf, 1280, 512, 1024);
  gemm_wmma<true,  false, false><<<dim3(1280 / 256, 512 / 64), 256, 0, stream>>>(
      Hf, Wf2b, bf2, fused, 1280, 512, 512);

  const float scale = 0.044194173824159216f;  // 1/sqrt(512)

  for (int blk = 0; blk < 3; ++blk) {
    const float* a5 = (blk == 0) ? word : (blk == 1) ? image : fused;
    build_full<<<(NBAT * SEQ * E_DIM + 255) / 256, 256, 0, stream>>>(Xf, a5, query);

    // Per-head QKV projections as (M*8, 64) x (64, 64) GEMMs, bf16 outputs
    const int Mh = (int)(MR * NH);  // 163840
    gemm_wmma<false, false, true><<<dim3(Mh / 256, 1), 256, 0, stream>>>(
        Xf, Wqb[blk], nullptr, Qb, Mh, 64, 64);
    gemm_wmma<false, false, true><<<dim3(Mh / 256, 1), 256, 0, stream>>>(
        Xf, Wkb[blk], nullptr, Kb, Mh, 64, 64);
    gemm_wmma<false, false, true><<<dim3(Mh / 256, 1), 256, 0, stream>>>(
        Xf, Wvb[blk], nullptr, Vb, Mh, 64, 64);

    attn_wmma<<<NBAT * NH, 160, 0, stream>>>(Qb, Kb, Vb, Obuf, scale);

    add_ln<<<(unsigned)(MR / 8), 256, 0, stream>>>(Xf, Obuf, ga[blk], ba[blk], Ybuf);

    gemm_wmma<false, true,  true ><<<dim3((unsigned)(MR / 256), 8), 256, 0, stream>>>(
        Ybuf, W1b[blk], b1[blk], H1, (int)MR, 512, 512);
    gemm_wmma<true,  false, false><<<dim3((unsigned)(MR / 256), 8), 256, 0, stream>>>(
        H1, W2b[blk], b2[blk], H2, (int)MR, 512, 512);

    add_ln<<<(unsigned)(MR / 8), 256, 0, stream>>>(Ybuf, H2, gn[blk], bn[blk], Xf);

    slice_out<<<(NBAT * 6 * E_DIM + 255) / 256, 256, 0, stream>>>(
        Xf, out + (size_t)blk * 655360, out + 1966080 + (size_t)blk * 131072);
  }
}
